// MultiScaleGATLayer_70308614636084
// MI455X (gfx1250) — matrix operand order repaired and checked by reading, not compiled, not run on hardware
//
#include <hip/hip_runtime.h>
#include <hip/hip_bf16.h>

#define DIM 128          // IN_DIM == OUT_DIM
#define HEADS 4
#define NEG_SLOPE 0.2f
#define LN_EPS 1e-5f

typedef unsigned short u16;
typedef __attribute__((ext_vector_type(16))) __bf16 v16bf;
typedef __attribute__((ext_vector_type(8)))  float  v8f;

union BfFrag { v16bf v; uint4 q[2]; };

__device__ __forceinline__ u16 f2bf(float f) {
    unsigned int u = __float_as_uint(f);
    u += 0x7FFFu + ((u >> 16) & 1u);          // round-to-nearest-even
    return (u16)(u >> 16);
}

// order-preserving float<->uint encoding for atomicMax on signed floats
__device__ __forceinline__ unsigned int enc_f32(float f) {
    unsigned int b = __float_as_uint(f);
    return (b & 0x80000000u) ? ~b : (b | 0x80000000u);
}
__device__ __forceinline__ float dec_f32(unsigned int u) {
    unsigned int b = (u & 0x80000000u) ? (u & 0x7FFFFFFFu) : ~u;
    return __uint_as_float(b);
}

// ---------------------------------------------------------------- init
__global__ void gat_init(float* __restrict__ acc, float* __restrict__ denom,
                         unsigned int* __restrict__ smax, size_t nAcc, size_t nSeg) {
    size_t i = (size_t)blockIdx.x * blockDim.x + threadIdx.x;
    size_t stride = (size_t)gridDim.x * blockDim.x;
    for (size_t j = i; j < nAcc; j += stride) acc[j] = 0.0f;
    for (size_t j = i; j < nSeg; j += stride) { denom[j] = 0.0f; smax[j] = 0u; }
}

// ------------------------------------------------------ one-time f32 -> bf16
__global__ void cvt_x_bf16(const float* __restrict__ x, u16* __restrict__ xb, size_t n4) {
    size_t i = (size_t)blockIdx.x * blockDim.x + threadIdx.x;
    size_t stride = (size_t)gridDim.x * blockDim.x;
    for (size_t j = i; j < n4; j += stride) {
        float4 f = ((const float4*)x)[j];
        uint2 p;
        p.x = (unsigned int)f2bf(f.x) | ((unsigned int)f2bf(f.y) << 16);
        p.y = (unsigned int)f2bf(f.z) | ((unsigned int)f2bf(f.w) << 16);
        ((uint2*)xb)[j] = p;
    }
}

// wt[c][k] = W[k][c] in bf16, c in [0,256): 0-127 -> Wl column, 128-255 -> Wr column
__global__ void cvt_w_bf16(const float* __restrict__ Wl, const float* __restrict__ Wr,
                           u16* __restrict__ wt) {
    int idx = blockIdx.x * blockDim.x + threadIdx.x;
    if (idx >= 256 * DIM) return;
    const int c = idx >> 7, k = idx & (DIM - 1);
    const float v = (c < DIM) ? Wl[(size_t)k * DIM + c] : Wr[(size_t)k * DIM + (c - DIM)];
    wt[idx] = f2bf(v);
}

// ------------------------------------------------- fused dual GEMM (WMMA bf16)
// [xl | xr] = x @ [Wl | Wr] + [bl | br]. One wave computes a 16x64 strip:
// 4 accumulators share one A fragment -> 16 v_wmma per wave, zero cvt in loop.
__global__ void gat_gemm_wmma(const u16* __restrict__ xb, const u16* __restrict__ wt,
                              const float* __restrict__ bl, const float* __restrict__ br,
                              float* __restrict__ xl, float* __restrict__ xr,
                              int nNodes) {
    const int wave = threadIdx.x >> 5;
    const int lane = threadIdx.x & 31;
    const int tile = blockIdx.x * 8 + wave;            // 8 waves / 256-thread block
    const int nRowTiles = (nNodes + 15) >> 4;
    if (tile >= nRowTiles * 4) return;

    const int rowTile = tile >> 2;
    const int cg      = tile & 3;                      // 64-col group in 256-wide output
    const int half = lane >> 4;                        // 0: lanes 0-15, 1: lanes 16-31
    const int l16  = lane & 15;
    const int row  = rowTile * 16 + l16;               // A row owned by this lane

    float*       dst = (cg < 2) ? xl : xr;
    const float* bb  = (cg < 2) ? bl : br;
    const int    n0  = (cg & 1) * 64;                  // base column within xl/xr

    v8f c0 = {}, c1 = {}, c2 = {}, c3 = {};
    #pragma unroll
    for (int kt = 0; kt < DIM; kt += 32) {
        // A fragment 16x32: lane half selects K {0..15} / {16..31}
        BfFrag a;
        {
            const u16* ap = xb + (size_t)row * DIM + kt + half * 16;
            a.q[0] = *(const uint4*)(ap);
            a.q[1] = *(const uint4*)(ap + 8);
        }
        // B fragments 32x16: lane owns column, K contiguous in transposed wt
        BfFrag b0, b1, b2, b3;
        {
            const size_t cbase = (size_t)(cg * 64 + l16) * DIM + kt + half * 16;
            const u16* bp = wt + cbase;
            b0.q[0] = *(const uint4*)(bp);       b0.q[1] = *(const uint4*)(bp + 8);
            bp += (size_t)16 * DIM;
            b1.q[0] = *(const uint4*)(bp);       b1.q[1] = *(const uint4*)(bp + 8);
            bp += (size_t)16 * DIM;
            b2.q[0] = *(const uint4*)(bp);       b2.q[1] = *(const uint4*)(bp + 8);
            bp += (size_t)16 * DIM;
            b3.q[0] = *(const uint4*)(bp);       b3.q[1] = *(const uint4*)(bp + 8);
        }
        c0 = __builtin_amdgcn_wmma_f32_16x16x32_bf16(false, a.v, false, b0.v, (short)0, c0, false, false);
        c1 = __builtin_amdgcn_wmma_f32_16x16x32_bf16(false, a.v, false, b1.v, (short)0, c1, false, false);
        c2 = __builtin_amdgcn_wmma_f32_16x16x32_bf16(false, a.v, false, b2.v, (short)0, c2, false, false);
        c3 = __builtin_amdgcn_wmma_f32_16x16x32_bf16(false, a.v, false, b3.v, (short)0, c3, false, false);
    }

    // D: VGPR j -> M = half*8 + j, N = l16 (per 16-wide subtile)
    const int rbase = rowTile * 16 + half * 8;
    #pragma unroll
    for (int sub = 0; sub < 4; ++sub) {
        const v8f& c = (sub == 0) ? c0 : (sub == 1) ? c1 : (sub == 2) ? c2 : c3;
        const int col = n0 + sub * 16 + l16;
        const float bc = bb[col];
        #pragma unroll
        for (int j = 0; j < 8; ++j)
            dst[(size_t)(rbase + j) * DIM + col] = c[j] + bc;
    }
}

// ---------------------------------------------------------------- edge scores
// One wave per edge; lane owns 4 dims, 8 lanes per head.
__global__ void gat_score(const int* __restrict__ ei, const float* __restrict__ ea,
                          const float* __restrict__ xl, const float* __restrict__ xr,
                          const float* __restrict__ We, const float* __restrict__ att,
                          float* __restrict__ score, unsigned int* __restrict__ smax,
                          int nEdges) {
    const int wave = threadIdx.x >> 5;
    const int lane = threadIdx.x & 31;
    const int e = blockIdx.x * 8 + wave;
    if (e >= nEdges) return;
    const int src = ei[e];
    const int dst = ei[nEdges + e];
    const int d0 = lane * 4;
    const int h  = lane >> 3;

    float4 ml = *(const float4*)(xl + (size_t)src * DIM + d0);   // L2-resident gather
    float4 mr = *(const float4*)(xr + (size_t)dst * DIM + d0);
    float4 we = *(const float4*)(We + d0);
    float4 at = *(const float4*)(att + d0);
    const float eav = ea[e];

    float p = 0.0f, m;
    m = ml.x + mr.x + eav * we.x; m = (m >= 0.0f) ? m : NEG_SLOPE * m; p += m * at.x;
    m = ml.y + mr.y + eav * we.y; m = (m >= 0.0f) ? m : NEG_SLOPE * m; p += m * at.y;
    m = ml.z + mr.z + eav * we.z; m = (m >= 0.0f) ? m : NEG_SLOPE * m; p += m * at.z;
    m = ml.w + mr.w + eav * we.w; m = (m >= 0.0f) ? m : NEG_SLOPE * m; p += m * at.w;

    #pragma unroll
    for (int off = 1; off < 8; off <<= 1) p += __shfl_xor(p, off, 32);

    if ((lane & 7) == 0) {
        score[(size_t)e * HEADS + h] = p;
        atomicMax(smax + (size_t)dst * HEADS + h, enc_f32(p));
    }
}

// --------------------------------------------------------- softmax + aggregate
__global__ void gat_aggregate(const int* __restrict__ ei, const float* __restrict__ xl,
                              const float* __restrict__ score, const unsigned int* __restrict__ smax,
                              float* __restrict__ denom, float* __restrict__ acc,
                              int nEdges) {
    const int wave = threadIdx.x >> 5;
    const int lane = threadIdx.x & 31;
    const int e = blockIdx.x * 8 + wave;
    if (e >= nEdges) return;
    const int src = ei[e];
    const int dst = ei[nEdges + e];
    const int d0 = lane * 4;
    const int h  = lane >> 3;

    const float s  = score[(size_t)e * HEADS + h];
    const float mx = dec_f32(smax[(size_t)dst * HEADS + h]);
    const float ex = __expf(s - mx);

    if ((lane & 7) == 0) atomicAdd(denom + (size_t)dst * HEADS + h, ex);

    float4 v = *(const float4*)(xl + (size_t)src * DIM + d0);
    float* o = acc + (size_t)dst * DIM + d0;
    atomicAdd(o + 0, ex * v.x);
    atomicAdd(o + 1, ex * v.y);
    atomicAdd(o + 2, ex * v.z);
    atomicAdd(o + 3, ex * v.w);
}

// ------------------------------------------------ bias + residual + LayerNorm
__global__ void gat_finalize(const float* __restrict__ x, const float* __restrict__ denom,
                             const float* __restrict__ bias, const float* __restrict__ gamma,
                             const float* __restrict__ beta, float* __restrict__ out,
                             int nNodes) {
    const int wave = threadIdx.x >> 5;
    const int lane = threadIdx.x & 31;
    const int n = blockIdx.x * 8 + wave;
    if (n >= nNodes) return;
    const int d0 = lane * 4;
    const int h  = lane >> 3;

    const float inv = 1.0f / (denom[(size_t)n * HEADS + h] + 1e-16f);
    float4 a  = *(const float4*)(out  + (size_t)n * DIM + d0);
    float4 xb = *(const float4*)(x    + (size_t)n * DIM + d0);
    float4 bi = *(const float4*)(bias + d0);

    float v0 = a.x * inv + bi.x + xb.x;
    float v1 = a.y * inv + bi.y + xb.y;
    float v2 = a.z * inv + bi.z + xb.z;
    float v3 = a.w * inv + bi.w + xb.w;

    float s  = v0 + v1 + v2 + v3;
    float sq = v0 * v0 + v1 * v1 + v2 * v2 + v3 * v3;
    #pragma unroll
    for (int off = 1; off < 32; off <<= 1) {
        s  += __shfl_xor(s,  off, 32);
        sq += __shfl_xor(sq, off, 32);
    }
    const float mean = s * (1.0f / DIM);
    const float var  = sq * (1.0f / DIM) - mean * mean;
    const float rstd = rsqrtf(var + LN_EPS);

    float4 g  = *(const float4*)(gamma + d0);
    float4 bt = *(const float4*)(beta  + d0);
    float4 o;
    o.x = (v0 - mean) * rstd * g.x + bt.x;
    o.y = (v1 - mean) * rstd * g.y + bt.y;
    o.z = (v2 - mean) * rstd * g.z + bt.z;
    o.w = (v3 - mean) * rstd * g.w + bt.w;
    *(float4*)(out + (size_t)n * DIM + d0) = o;
}

extern "C" void kernel_launch(void* const* d_in, const int* in_sizes, int n_in,
                              void* d_out, int out_size, void* d_ws, size_t ws_size,
                              hipStream_t stream) {
    const float* x    = (const float*)d_in[0];
    const int*   ei   = (const int*)d_in[1];
    const float* ea   = (const float*)d_in[2];
    const float* Wl   = (const float*)d_in[3];
    const float* bl   = (const float*)d_in[4];
    const float* Wr   = (const float*)d_in[5];
    const float* br   = (const float*)d_in[6];
    const float* We   = (const float*)d_in[7];
    const float* att  = (const float*)d_in[8];
    const float* bias = (const float*)d_in[9];
    const float* gam  = (const float*)d_in[10];
    const float* bet  = (const float*)d_in[11];

    const int nNodes = in_sizes[0] / DIM;
    const int nEdges = in_sizes[1] / 2;

    // workspace: xl | xr | score | smax | denom | xb(bf16) | wt(bf16)   (~78.5 MB here)
    float* xl = (float*)d_ws;
    float* xr = xl + (size_t)nNodes * DIM;
    float* score = xr + (size_t)nNodes * DIM;
    unsigned int* smax = (unsigned int*)(score + (size_t)nEdges * HEADS);
    float* denom = (float*)(smax + (size_t)nNodes * HEADS);
    u16* xb = (u16*)(denom + (size_t)nNodes * HEADS);
    u16* wt = xb + (size_t)nNodes * DIM;
    float* out = (float*)d_out;            // also used as the aggregation accumulator

    gat_init<<<2048, 256, 0, stream>>>(out, denom, smax,
                                       (size_t)nNodes * DIM, (size_t)nNodes * HEADS);

    cvt_x_bf16<<<2048, 256, 0, stream>>>(x, xb, (size_t)nNodes * DIM / 4);
    cvt_w_bf16<<<(256 * DIM + 255) / 256, 256, 0, stream>>>(Wl, Wr, wt);

    const int nTiles = ((nNodes + 15) / 16) * 4;       // row tiles x 4 column groups
    gat_gemm_wmma<<<(nTiles + 7) / 8, 256, 0, stream>>>(xb, wt, bl, br, xl, xr, nNodes);

    gat_score<<<(nEdges + 7) / 8, 256, 0, stream>>>(ei, ea, xl, xr, We, att,
                                                    score, smax, nEdges);

    gat_aggregate<<<(nEdges + 7) / 8, 256, 0, stream>>>(ei, xl, score, smax,
                                                        denom, out, nEdges);

    gat_finalize<<<(nNodes + 7) / 8, 256, 0, stream>>>(x, denom, bias, gam, bet,
                                                       out, nNodes);
}
